// GNN_49478023250692
// MI455X (gfx1250) — matrix operand order repaired and checked by reading, not compiled, not run on hardware
//
#include <hip/hip_runtime.h>

typedef unsigned short u16;
typedef __attribute__((ext_vector_type(16))) __bf16 v16bf;
typedef __attribute__((ext_vector_type(8)))  float  v8f;

#define Nn   100000
#define En   800000
#define E3n  400000
#define DIMn 128

// ---------- helpers ----------
__device__ __forceinline__ u16 f2bf(float f) {
  union { float f; unsigned int u; } x; x.f = f;
  unsigned int r = x.u + 0x7FFFu + ((x.u >> 16) & 1u);
  return (u16)(r >> 16);
}

// A fragment: 16x32 bf16, row-major source with leading dim `lda` (elements).
// lane 0-15: row = lane,   K = k0+0..7  and k0+16..23
// lane 16-31: row = lane-16, K = k0+8..15 and k0+24..31
__device__ __forceinline__ v16bf load_a_frag(const u16* A, int lda, int row0, int k0, int lane) {
  int half = lane >> 4;
  int r = row0 + (lane & 15);
  const u16* p = A + (size_t)r * lda + k0 + half * 8;
  union { uint4 u[2]; v16bf v; } U;
  U.u[0] = *(const uint4*)(p);
  U.u[1] = *(const uint4*)(p + 16);
  return U.v;
}

// B fragment from pre-swizzled weights: Wp[((kt*nTiles+nt)*32+lane)*16 + j]
__device__ __forceinline__ v16bf load_b_frag(const u16* Wp, int kt, int nt, int nTiles, int lane) {
  const u16* p = Wp + (((size_t)kt * nTiles + nt) * 32 + lane) * 16;
  union { uint4 u[2]; v16bf v; } U;
  U.u[0] = *(const uint4*)(p);
  U.u[1] = *(const uint4*)(p + 8);
  return U.v;
}

// ---------- elementwise / stats kernels ----------
__global__ void zerof(float* p, long long n) {
  long long t = (long long)blockIdx.x * blockDim.x + threadIdx.x;
  if (t < n) p[t] = 0.f;
}

// per-column sum and sum-of-squares; launch C blocks of 256 (R=256 row slices)
__global__ void colstats(const float* __restrict__ X, int Nrows, int C, float* __restrict__ sums) {
  int t = blockIdx.x * blockDim.x + threadIdx.x;
  if (t >= C * 256) return;
  int f = t % C, rs = t / C;
  float s = 0.f, sq = 0.f;
  for (int r = rs; r < Nrows; r += 256) {
    float v = X[(size_t)r * C + f];
    s += v; sq += v * v;
  }
  atomicAdd(&sums[f], s);
  atomicAdd(&sums[C + f], sq);
}

__global__ void bn_finalize(const float* sums, const float* g, const float* b,
                            int C, float Ninv, float* scale, float* shift) {
  int f = blockIdx.x * blockDim.x + threadIdx.x;
  if (f >= C) return;
  float m = sums[f] * Ninv;
  float var = sums[C + f] * Ninv - m * m;
  float sc = g[f] * rsqrtf(var + 1e-5f);
  scale[f] = sc;
  shift[f] = b[f] - m * sc;
}

// out[N,Cpad] = bf16(X*scale+shift), zero-padded beyond C
__global__ void bn_apply_bf(const float* __restrict__ X, const float* scale, const float* shift,
                            int Nrows, int C, int Cpad, u16* __restrict__ out) {
  size_t t = (size_t)blockIdx.x * blockDim.x + threadIdx.x;
  size_t total = (size_t)Nrows * Cpad;
  if (t >= total) return;
  int c = (int)(t % Cpad);
  size_t r = t / Cpad;
  u16 v = 0;
  if (c < C) v = f2bf(X[r * C + c] * scale[c] + shift[c]);
  out[t] = v;
}

// swizzle W[Kact,Ncols] f32 -> wave32 B-fragment bf16 layout, K padded to Kpad
__global__ void prep_weights(const float* __restrict__ W, u16* __restrict__ Wp,
                             int Kact, int Kpad, int Ncols) {
  int t = blockIdx.x * blockDim.x + threadIdx.x;
  int total = Kpad * Ncols;
  if (t >= total) return;
  int j = t & 15;
  int lane = (t >> 4) & 31;
  int tile = t >> 9;
  int nTiles = Ncols >> 4;
  int kt = tile / nTiles, nt = tile % nTiles;
  int k = kt * 32 + ((lane >> 4) << 3) + (j & 7) + ((j >> 3) << 4);
  int n = nt * 16 + (lane & 15);
  u16 v = 0;
  if (k < Kact) v = f2bf(W[(size_t)k * Ncols + n]);
  Wp[t] = v;
}

// ---------- WMMA GEMM: out = act( A0@W0 [+ A1@W1] + bias ) ----------
__global__ void wmma_gemm(const u16* __restrict__ A0, const u16* __restrict__ Wp0, int kt0,
                          const u16* __restrict__ A1, const u16* __restrict__ Wp1, int kt1,
                          const float* __restrict__ bias, int relu,
                          float* __restrict__ outF, u16* __restrict__ outB,
                          int M, int Ncols) {
  int nTiles = Ncols >> 4;
  int mTiles = M >> 4;
  int gid = blockIdx.x * (blockDim.x >> 5) + (threadIdx.x >> 5);
  if (gid >= mTiles * nTiles) return;   // wave-uniform
  int lane = threadIdx.x & 31;
  int mt = gid / nTiles, nt = gid % nTiles;
  int row0 = mt << 4;
  v8f c = {};
  int lda0 = kt0 << 5;
  for (int kt = 0; kt < kt0; ++kt) {
    v16bf a = load_a_frag(A0, lda0, row0, kt << 5, lane);
    v16bf b = load_b_frag(Wp0, kt, nt, nTiles, lane);
    c = __builtin_amdgcn_wmma_f32_16x16x32_bf16(false, a, false, b, (short)0, c, false, false);
  }
  if (A1) {
    int lda1 = kt1 << 5;
    for (int kt = 0; kt < kt1; ++kt) {
      v16bf a = load_a_frag(A1, lda1, row0, kt << 5, lane);
      v16bf b = load_b_frag(Wp1, kt, nt, nTiles, lane);
      c = __builtin_amdgcn_wmma_f32_16x16x32_bf16(false, a, false, b, (short)0, c, false, false);
    }
  }
  int half = lane >> 4, nl = lane & 15, n = (nt << 4) + nl;
  float bv = bias ? bias[n] : 0.f;
#pragma unroll
  for (int i = 0; i < 8; ++i) {
    int m = row0 + i + (half << 3);
    float v = c[i] + bv;
    if (relu) v = fmaxf(v, 0.f);
    if (outF) outF[(size_t)m * Ncols + n] = v;
    if (outB) outB[(size_t)m * Ncols + n] = f2bf(v);
  }
}

// ---------- graph kernels ----------
__global__ void count_deg(const int* __restrict__ dstArr, float* __restrict__ cnt, int E) {
  int t = blockIdx.x * blockDim.x + threadIdx.x;
  if (t < E) atomicAdd(&cnt[dstArr[t]], 1.f);
}

__global__ void count_deg_sym(const int* __restrict__ ei3, float* __restrict__ cnt, int E3) {
  int t = blockIdx.x * blockDim.x + threadIdx.x;
  if (t >= 2 * E3) return;
  int d = (t < E3) ? ei3[E3 + t] : ei3[t - E3];
  atomicAdd(&cnt[d], 1.f);
}

// wave per edge: msg = relu(hs[src]+hd[dst]+ea@We+bm); atomic scatter to aggr[dst]
__global__ void edge_msg(int mode, const int* __restrict__ ei,
                         const float* __restrict__ ea,
                         const float* __restrict__ ea3, const float* __restrict__ ea4,
                         const float* __restrict__ We, const float* __restrict__ bm, int KE,
                         const float* __restrict__ hs, const float* __restrict__ hd,
                         float* __restrict__ aggr, int Etot, int Ehalf) {
  __shared__ float sWe[16 * 128];
  __shared__ float sBm[128];
  for (int i = threadIdx.x; i < KE * 128; i += blockDim.x) sWe[i] = We[i];
  for (int i = threadIdx.x; i < 128; i += blockDim.x) sBm[i] = bm[i];
  __syncthreads();
  int lane = threadIdx.x & 31;
  int wid = blockIdx.x * (blockDim.x >> 5) + (threadIdx.x >> 5);
  int nw = gridDim.x * (blockDim.x >> 5);
  float ev[16];
  for (int e = wid; e < Etot; e += nw) {
    int s, d;
    if (mode == 0) {
      s = ei[e]; d = ei[Etot + e];
      for (int k = 0; k < 8; ++k) ev[k] = ea[(size_t)e * 8 + k];
    } else {
      int r = (e < Ehalf) ? e : e - Ehalf;
      if (e < Ehalf) { s = ei[r]; d = ei[Ehalf + r]; }
      else           { s = ei[Ehalf + r]; d = ei[r]; }
      for (int k = 0; k < 8; ++k) { ev[k] = ea3[(size_t)r * 8 + k]; ev[8 + k] = ea4[(size_t)r * 8 + k]; }
    }
    size_t so = (size_t)s * 128, dofs = (size_t)d * 128;
#pragma unroll
    for (int j = 0; j < 4; ++j) {
      int f = lane + (j << 5);
      float acc = sBm[f];
      for (int k = 0; k < KE; ++k) acc += ev[k] * sWe[k * 128 + f];
      acc += hs[so + f] + hd[dofs + f];
      if (acc > 0.f) atomicAdd(&aggr[dofs + f], acc);
    }
  }
}

// aggr_bf = bf16(aggr / max(count,1))
__global__ void div_convert(const float* __restrict__ aggr, const float* __restrict__ cnt,
                            u16* __restrict__ out, int Nrows) {
  size_t t = (size_t)blockIdx.x * blockDim.x + threadIdx.x;
  if (t >= (size_t)Nrows * 128) return;
  int n = (int)(t >> 7);
  float inv = 1.f / fmaxf(cnt[n], 1.f);
  out[t] = f2bf(aggr[t] * inv);
}

// ---------- head: feat[16,288] tile per wave, WMMA 272->128 + ReLU + 128->1 ----------
__global__ void head_kernel(const u16* __restrict__ hbf, const int* __restrict__ ei3,
                            const float* __restrict__ ea3, const float* __restrict__ ea4,
                            const u16* __restrict__ Wp, const float* __restrict__ b1,
                            const float* __restrict__ W2, const float* __restrict__ b2,
                            float* __restrict__ out, int E3) {
  __shared__ alignas(16) u16 feat[2][16 * 288];
  int w = threadIdx.x >> 5, lane = threadIdx.x & 31;
  int gw = blockIdx.x * 2 + w;
  int e0 = gw << 4;
  u16* fr = feat[w];
  for (int i = lane; i < 16 * 288; i += 32) {
    int ed = i / 288, cc = i - ed * 288;
    int e = e0 + ed;
    u16 v = 0;
    if (cc < 128)      { int s = ei3[e];      v = hbf[(size_t)s * 128 + cc]; }
    else if (cc < 256) { int d = ei3[E3 + e]; v = hbf[(size_t)d * 128 + (cc - 128)]; }
    else if (cc < 264) v = f2bf(ea3[(size_t)e * 8 + (cc - 256)]);
    else if (cc < 272) v = f2bf(ea4[(size_t)e * 8 + (cc - 264)]);
    fr[i] = v;
  }
  __syncthreads();
  float scal[8] = {0, 0, 0, 0, 0, 0, 0, 0};
  int nl = lane & 15;
  for (int ntl = 0; ntl < 8; ++ntl) {
    v8f c = {};
    for (int kt = 0; kt < 9; ++kt) {
      v16bf a = load_a_frag(fr, 288, 0, kt << 5, lane);
      v16bf b = load_b_frag(Wp, kt, ntl, 8, lane);
      c = __builtin_amdgcn_wmma_f32_16x16x32_bf16(false, a, false, b, (short)0, c, false, false);
    }
    int n = (ntl << 4) + nl;
    float bb = b1[n], w2 = W2[n];
#pragma unroll
    for (int i = 0; i < 8; ++i) scal[i] += fmaxf(c[i] + bb, 0.f) * w2;
  }
  float bias2 = b2[0];
#pragma unroll
  for (int i = 0; i < 8; ++i)
    for (int m = 1; m < 16; m <<= 1) scal[i] += __shfl_xor(scal[i], m, 32);
  if (lane == 0)  for (int i = 0; i < 8; ++i) out[e0 + i]     = scal[i] + bias2;
  if (lane == 16) for (int i = 0; i < 8; ++i) out[e0 + 8 + i] = scal[i] + bias2;
}

// ---------- host ----------
extern "C" void kernel_launch(void* const* d_in, const int* in_sizes, int n_in,
                              void* d_out, int out_size, void* d_ws, size_t ws_size,
                              hipStream_t stream) {
  const float* x      = (const float*)d_in[0];
  const int*   ei     = (const int*)d_in[1];
  const float* ea     = (const float*)d_in[2];
  const int*   ei3    = (const int*)d_in[3];
  const float* ea3    = (const float*)d_in[4];
  const float* ea4    = (const float*)d_in[5];
  const float* bn1_g  = (const float*)d_in[6];
  const float* bn1_b  = (const float*)d_in[7];
  const float* lin1_W = (const float*)d_in[8];
  const float* lin1_b = (const float*)d_in[9];
  const float* bn2_g  = (const float*)d_in[10];
  const float* bn2_b  = (const float*)d_in[11];
  const float* lin2_W = (const float*)d_in[12];
  const float* lin2_b = (const float*)d_in[13];
  const float* c1_Ws  = (const float*)d_in[14];
  const float* c1_Wd  = (const float*)d_in[15];
  const float* c1_We  = (const float*)d_in[16];
  const float* c1_bm  = (const float*)d_in[17];
  const float* c1_Wu  = (const float*)d_in[18];
  const float* c1_bu  = (const float*)d_in[19];
  const float* c2_Ws  = (const float*)d_in[20];
  const float* c2_Wd  = (const float*)d_in[21];
  const float* c2_We  = (const float*)d_in[22];
  const float* c2_bm  = (const float*)d_in[23];
  const float* c2_Wu  = (const float*)d_in[24];
  const float* c2_bu  = (const float*)d_in[25];
  const float* hW1    = (const float*)d_in[26];
  const float* hb1    = (const float*)d_in[27];
  const float* hW2    = (const float*)d_in[28];
  const float* hb2    = (const float*)d_in[29];
  float* out = (float*)d_out;

  char* ws = (char*)d_ws;
  size_t o = 0;
  auto alloc = [&](size_t bytes) { size_t r = o; o += (bytes + 255) & ~(size_t)255; return r; };

  size_t rA   = alloc((size_t)Nn * 256 * 4);   // h256  / later hs(0..)+hd(+N*128)
  size_t rB   = alloc((size_t)Nn * 256 * 2);   // h256_bf / later aggr(f32, same size)
  size_t rHB0 = alloc((size_t)Nn * 128 * 2);   // bf16 activations ping
  size_t rHB1 = alloc((size_t)Nn * 128 * 2);   // bf16 activations pong
  size_t rAGB = alloc((size_t)Nn * 128 * 2);   // aggr bf16
  size_t rXB  = alloc((size_t)Nn * 32 * 2);    // padded bf16 input
  size_t rCNT = alloc((size_t)Nn * 4);         // in-degree counts
  size_t rST  = alloc(2048 * 4);               // bn stats: sums[512], scale[256], shift[256]
  size_t rW0  = alloc(131072);
  size_t rW1  = alloc(131072);
  size_t rW2  = alloc(131072);
  size_t rW3  = alloc(131072);

  float* h256   = (float*)(ws + rA);
  float* hs     = (float*)(ws + rA);
  float* hd     = (float*)(ws + rA) + (size_t)Nn * 128;
  u16*   h256bf = (u16*)(ws + rB);
  float* aggr   = (float*)(ws + rB);
  u16*   hb[2]  = { (u16*)(ws + rHB0), (u16*)(ws + rHB1) };
  u16*   aggrbf = (u16*)(ws + rAGB);
  u16*   xbf    = (u16*)(ws + rXB);
  float* cnt    = (float*)(ws + rCNT);
  float* sums   = (float*)(ws + rST);
  float* scale  = sums + 512;
  float* shift  = sums + 768;
  u16*   wp0    = (u16*)(ws + rW0);
  u16*   wp1    = (u16*)(ws + rW1);
  u16*   wp2    = (u16*)(ws + rW2);
  u16*   wp3    = (u16*)(ws + rW3);

  auto cdiv = [](long long a, long long b) { return (int)((a + b - 1) / b); };
  auto gemm = [&](const u16* A0, const u16* W0, int kt0, const u16* A1, const u16* W1, int kt1,
                  const float* bias, int relu, float* oF, u16* oB, int M, int Ncols) {
    int waves = (M / 16) * (Ncols / 16);
    wmma_gemm<<<cdiv(waves, 8), 256, 0, stream>>>(A0, W0, kt0, A1, W1, kt1, bias, relu, oF, oB, M, Ncols);
  };
  auto prep = [&](const float* W, u16* Wp, int Kact, int Kpad, int Ncols) {
    prep_weights<<<cdiv((long long)Kpad * Ncols, 256), 256, 0, stream>>>(W, Wp, Kact, Kpad, Ncols);
  };
  float Ninv = 1.0f / (float)Nn;

  // ---- lin_node: BN1 -> lin1 -> relu -> BN2 -> lin2 -> relu ----
  zerof<<<1, 256, 0, stream>>>(sums, 512);
  colstats<<<13, 256, 0, stream>>>(x, Nn, 13, sums);
  bn_finalize<<<1, 32, 0, stream>>>(sums, bn1_g, bn1_b, 13, Ninv, scale, shift);
  bn_apply_bf<<<cdiv((long long)Nn * 32, 256), 256, 0, stream>>>(x, scale, shift, Nn, 13, 32, xbf);
  prep(lin1_W, wp0, 13, 32, 256);
  gemm(xbf, wp0, 1, nullptr, nullptr, 0, lin1_b, 1, h256, nullptr, Nn, 256);

  zerof<<<2, 256, 0, stream>>>(sums, 512);
  colstats<<<256, 256, 0, stream>>>(h256, Nn, 256, sums);
  bn_finalize<<<1, 256, 0, stream>>>(sums, bn2_g, bn2_b, 256, Ninv, scale, shift);
  bn_apply_bf<<<cdiv((long long)Nn * 256, 256), 256, 0, stream>>>(h256, scale, shift, Nn, 256, 256, h256bf);
  prep(lin2_W, wp0, 256, 256, 128);
  int cur = 0;
  gemm(h256bf, wp0, 8, nullptr, nullptr, 0, lin2_b, 1, nullptr, hb[cur], Nn, 128);

  long long nfeat = (long long)Nn * 128;

  // ---- conv1 stack (bond graph) ----
  zerof<<<cdiv(Nn, 256), 256, 0, stream>>>(cnt, Nn);
  count_deg<<<cdiv(En, 256), 256, 0, stream>>>(ei + En, cnt, En);
  for (int l = 0; l < 3; ++l) {
    prep(c1_Ws + (size_t)l * 128 * 128, wp0, 128, 128, 128);
    prep(c1_Wd + (size_t)l * 128 * 128, wp1, 128, 128, 128);
    prep(c1_Wu + (size_t)l * 256 * 128, wp2, 128, 128, 128);                   // rows 0..127 (h part)
    prep(c1_Wu + (size_t)l * 256 * 128 + 128 * 128, wp3, 128, 128, 128);       // rows 128..255 (aggr part)
    gemm(hb[cur], wp0, 4, nullptr, nullptr, 0, nullptr, 0, hs, nullptr, Nn, 128);
    gemm(hb[cur], wp1, 4, nullptr, nullptr, 0, nullptr, 0, hd, nullptr, Nn, 128);
    zerof<<<cdiv(nfeat, 256), 256, 0, stream>>>(aggr, nfeat);
    edge_msg<<<4096, 256, 0, stream>>>(0, ei, ea, nullptr, nullptr,
                                       c1_We + (size_t)l * 8 * 128, c1_bm + (size_t)l * 128, 8,
                                       hs, hd, aggr, En, 0);
    div_convert<<<cdiv(nfeat, 256), 256, 0, stream>>>(aggr, cnt, aggrbf, Nn);
    gemm(hb[cur], wp2, 4, aggrbf, wp3, 4, c1_bu + (size_t)l * 128, 1, nullptr, hb[1 - cur], Nn, 128);
    cur = 1 - cur;
  }

  // ---- conv2 stack (symmetrized coupling graph) ----
  zerof<<<cdiv(Nn, 256), 256, 0, stream>>>(cnt, Nn);
  count_deg_sym<<<cdiv(2 * E3n, 256), 256, 0, stream>>>(ei3, cnt, E3n);
  for (int l = 0; l < 3; ++l) {
    prep(c2_Ws + (size_t)l * 128 * 128, wp0, 128, 128, 128);
    prep(c2_Wd + (size_t)l * 128 * 128, wp1, 128, 128, 128);
    prep(c2_Wu + (size_t)l * 256 * 128, wp2, 128, 128, 128);
    prep(c2_Wu + (size_t)l * 256 * 128 + 128 * 128, wp3, 128, 128, 128);
    gemm(hb[cur], wp0, 4, nullptr, nullptr, 0, nullptr, 0, hs, nullptr, Nn, 128);
    gemm(hb[cur], wp1, 4, nullptr, nullptr, 0, nullptr, 0, hd, nullptr, Nn, 128);
    zerof<<<cdiv(nfeat, 256), 256, 0, stream>>>(aggr, nfeat);
    edge_msg<<<4096, 256, 0, stream>>>(1, ei3, nullptr, ea3, ea4,
                                       c2_We + (size_t)l * 16 * 128, c2_bm + (size_t)l * 128, 16,
                                       hs, hd, aggr, 2 * E3n, E3n);
    div_convert<<<cdiv(nfeat, 256), 256, 0, stream>>>(aggr, cnt, aggrbf, Nn);
    gemm(hb[cur], wp2, 4, aggrbf, wp3, 4, c2_bu + (size_t)l * 128, 1, nullptr, hb[1 - cur], Nn, 128);
    cur = 1 - cur;
  }

  // ---- head ----
  prep(hW1, wp0, 272, 288, 128);
  head_kernel<<<E3n / 32, 64, 0, stream>>>(hb[cur], ei3, ea3, ea4, wp0, hb1, hW2, hb2, out, E3n);
}
